// GraphEmbedding_89008902242794
// MI455X (gfx1250) — matrix operand order repaired and checked by reading, not compiled, not run on hardware
//
#include <hip/hip_runtime.h>
#include <stdint.h>

// ---------------- problem constants (match reference) ----------------
#define N_NODES 50000
#define N_EDGES 800000
#define ET      (N_EDGES + N_NODES)   // with self-loops = 850000
#define G_GR    64
#define HEADS   4
#define CH      64
#define HID     64                    // hidden feature dim (H*C averaged)
#define HC      256                   // HEADS*CH
#define F_IN    16
#define D_OUT   32
#define NEG_SLOPE 0.2f

typedef __bf16 bf16;
typedef __attribute__((ext_vector_type(16))) __bf16 v16bf;
typedef __attribute__((ext_vector_type(8)))  float  v8f;

union TileRegs { v16bf v; uint32_t u[8]; };

// ---------------- helpers ----------------
__device__ __forceinline__ float lrelu(float x) { return x > 0.f ? x : NEG_SLOPE * x; }

__device__ __forceinline__ void get_edge(const int* __restrict__ ei, int e, int& s, int& d) {
    if (e < N_EDGES) { s = ei[e]; d = ei[N_EDGES + e]; }
    else             { s = e - N_EDGES; d = s; }           // appended self-loops
}

// float atomic max via sign-aware integer punning (emax initialized to -inf bits)
__device__ __forceinline__ void atomic_max_f32(float* addr, float val) {
    if (val >= 0.f) atomicMax((int*)addr, __float_as_int(val));
    else            atomicMin((unsigned int*)addr, (unsigned int)__float_as_int(val));
}

// ---------------- generic fill ----------------
__global__ void __launch_bounds__(256) fill_u32_kernel(uint32_t* p, uint32_t v, int n) {
    int i = blockIdx.x * blockDim.x + threadIdx.x;
    if (i < n) p[i] = v;
}

// ---------------- input projection: h = relu(x@Wp + bp) ----------------
__global__ void __launch_bounds__(256)
proj_kernel(const float* __restrict__ x, const float* __restrict__ Wp,
            const float* __restrict__ bp, float* __restrict__ h, bf16* __restrict__ hb) {
    int i = blockIdx.x * blockDim.x + threadIdx.x;       // n*64 + j
    if (i >= N_NODES * HID) return;
    int n = i >> 6, j = i & 63;
    const float* xr = x + n * F_IN;
    float acc = bp[j];
#pragma unroll
    for (int k = 0; k < F_IN; ++k) acc += xr[k] * Wp[k * HID + j];
    acc = fmaxf(acc, 0.f);
    h[i] = acc;
    hb[i] = (bf16)acc;
}

// ---------------- W [64,256] f32 -> Wt [256,64] bf16 (col-major for B tiles) ------
__global__ void __launch_bounds__(256)
wt_kernel(const float* __restrict__ W, bf16* __restrict__ Wt) {
    int i = blockIdx.x * blockDim.x + threadIdx.x;       // k*256 + col
    if (i >= HID * HC) return;
    int k = i >> 8, col = i & 255;
    Wt[col * HID + k] = (bf16)W[i];
}

// ---------------- hw = h @ W  via v_wmma_f32_16x16x32_bf16 ----------------
// grid: (3125 M-tiles, 4 N-groups), 128 threads = 4 waves, 1 16x16 tile per wave
__global__ void __launch_bounds__(128)
gemm_kernel(const bf16* __restrict__ hb, const bf16* __restrict__ Wt, float* __restrict__ hw) {
    const int lane  = threadIdx.x & 31;
    const int wave  = threadIdx.x >> 5;
    const int m0    = blockIdx.x * 16;                   // node-row base (50000 = 3125*16, exact)
    const int ntile = blockIdx.y * 4 + wave;             // 0..15 -> cols 0..255
    const int half  = lane >> 4;
    const int mrowA = m0 + (lane & 15);                  // A-matrix row for this lane
    const int ncol  = ntile * 16 + (lane & 15);          // B/D column for this lane

    v8f acc = {};
#pragma unroll
    for (int t = 0; t < 2; ++t) {                        // K = 64 in two K=32 steps
        TileRegs a, b;
        const int kb = t * 32;
#pragma unroll
        for (int v = 0; v < 8; ++v) {
            // A 16x32 bf16 layout: VGPR v holds K pair at 16*(v>=4) + 8*half + 2*(v&3)
            int ka = kb + ((v >= 4) ? 16 : 0) + half * 8 + (v & 3) * 2;
            a.u[v] = *(const uint32_t*)(hb + mrowA * HID + ka);
            // B 32x16 bf16 layout: VGPR v holds K pair at 16*(v>=4) + 8*half + 2*(v&3) for column ncol
            int kbv = kb + ((v >= 4) ? 16 : 0) + half * 8 + (v & 3) * 2;
            b.u[v] = *(const uint32_t*)(Wt + ncol * HID + kbv);
        }
        acc = __builtin_amdgcn_wmma_f32_16x16x32_bf16(
            false, a.v, false, b.v, (short)0, acc, false, false);
    }
#pragma unroll
    for (int r = 0; r < 8; ++r) {                        // D: VGPR r -> M = r + 8*half
        int mrow = m0 + r + 8 * half;
        hw[mrow * HC + ncol] = acc[r];
    }
}

// ---------------- per-node attention logits ----------------
__global__ void __launch_bounds__(256)
logits_kernel(const float* __restrict__ hw, const float* __restrict__ a_s,
              const float* __restrict__ a_d, float* __restrict__ al_s, float* __restrict__ al_d) {
    int i = blockIdx.x * blockDim.x + threadIdx.x;       // n*4 + head
    if (i >= N_NODES * HEADS) return;
    int n = i >> 2, hh = i & 3;
    const float* p = hw + n * HC + hh * CH;
    float s = 0.f, d = 0.f;
#pragma unroll 8
    for (int c = 0; c < CH; ++c) { float v = p[c]; s += v * a_s[hh * CH + c]; d += v * a_d[hh * CH + c]; }
    al_s[i] = s; al_d[i] = d;
}

// ---------------- edge pass 1: segment max over dst ----------------
__global__ void __launch_bounds__(256)
edge_max_kernel(const int* __restrict__ ei, const float* __restrict__ al_s,
                const float* __restrict__ al_d, float* __restrict__ emax) {
    int e = blockIdx.x * blockDim.x + threadIdx.x;
    if (e >= ET) return;
    int s, d; get_edge(ei, e, s, d);
#pragma unroll
    for (int hh = 0; hh < HEADS; ++hh) {
        float v = lrelu(al_s[s * 4 + hh] + al_d[d * 4 + hh]);
        atomic_max_f32(&emax[d * 4 + hh], v);
    }
}

// ---------------- edge pass 2: exp + segment sum ----------------
__global__ void __launch_bounds__(256)
edge_exp_kernel(const int* __restrict__ ei, const float* __restrict__ al_s,
                const float* __restrict__ al_d, const float* __restrict__ emax,
                float* __restrict__ ex, float* __restrict__ denom) {
    int e = blockIdx.x * blockDim.x + threadIdx.x;
    if (e >= ET) return;
    int s, d; get_edge(ei, e, s, d);
#pragma unroll
    for (int hh = 0; hh < HEADS; ++hh) {
        float v  = lrelu(al_s[s * 4 + hh] + al_d[d * 4 + hh]);
        float xv = __expf(v - emax[d * 4 + hh]);
        ex[e * 4 + hh] = xv;
        atomicAdd(&denom[d * 4 + hh], xv);
    }
}

// ---------------- edge pass 3: alpha-weighted scatter (64 threads / edge) --------
__global__ void __launch_bounds__(256)
edge_scatter_kernel(const int* __restrict__ ei, const float* __restrict__ ex,
                    const float* __restrict__ denom, const float* __restrict__ hw,
                    float* __restrict__ accum) {
    int gid = blockIdx.x * blockDim.x + threadIdx.x;
    int e = gid >> 6;
    int c = gid & 63;
    if (e >= ET) return;
    int s, d; get_edge(ei, e, s, d);
#pragma unroll
    for (int hh = 0; hh < HEADS; ++hh) {
        float alpha = ex[e * 4 + hh] / denom[d * 4 + hh];
        atomicAdd(&accum[d * HC + hh * CH + c], alpha * hw[s * HC + hh * CH + c]);
    }
}

// ---------------- finalize: head mean + bias + relu, re-emit bf16 ----------------
__global__ void __launch_bounds__(256)
finalize_kernel(const float* __restrict__ accum, const float* __restrict__ b,
                float* __restrict__ h, bf16* __restrict__ hb) {
    int i = blockIdx.x * blockDim.x + threadIdx.x;       // n*64 + j
    if (i >= N_NODES * HID) return;
    int n = i >> 6, j = i & 63;
    const float* p = accum + n * HC + j;
    float m = (p[0] + p[64] + p[128] + p[192]) * 0.25f + b[j];
    m = fmaxf(m, 0.f);
    h[i] = m;
    hb[i] = (bf16)m;
}

// ---------------- global mean pool (atomics) ----------------
__global__ void __launch_bounds__(256)
pool_kernel(const float* __restrict__ h, const int* __restrict__ batch,
            float* __restrict__ pooled, float* __restrict__ cnt) {
    int i = blockIdx.x * blockDim.x + threadIdx.x;       // n*64 + j
    if (i >= N_NODES * HID) return;
    int n = i >> 6, j = i & 63;
    int g = batch[n];
    atomicAdd(&pooled[g * HID + j], h[i]);
    if (j == 0) atomicAdd(&cnt[g], 1.f);
}

// ---------------- output head: (pooled/cnt) @ Wo + bo ----------------
__global__ void __launch_bounds__(256)
out_kernel(const float* __restrict__ pooled, const float* __restrict__ cnt,
           const float* __restrict__ Wo, const float* __restrict__ bo,
           float* __restrict__ out) {
    int i = blockIdx.x * blockDim.x + threadIdx.x;       // g*32 + d
    if (i >= G_GR * D_OUT) return;
    int g = i >> 5, dd = i & 31;
    float inv = 1.f / fmaxf(cnt[g], 1.f);
    float acc = bo[dd];
#pragma unroll 8
    for (int j = 0; j < HID; ++j) acc += (pooled[g * HID + j] * inv) * Wo[j * D_OUT + dd];
    out[i] = acc;
}

// ---------------- host orchestration ----------------
extern "C" void kernel_launch(void* const* d_in, const int* in_sizes, int n_in,
                              void* d_out, int out_size, void* d_ws, size_t ws_size,
                              hipStream_t stream) {
    (void)in_sizes; (void)n_in; (void)out_size; (void)ws_size;

    const float* x   = (const float*)d_in[0];
    const int*   ei  = (const int*)  d_in[1];
    const int*   bat = (const int*)  d_in[2];
    const float* Wp  = (const float*)d_in[3];
    const float* bp  = (const float*)d_in[4];
    const float* Wo  = (const float*)d_in[5];
    const float* bo  = (const float*)d_in[6];

    // workspace carve-out (≈138 MB, L2-resident on MI455X)
    char* ws = (char*)d_ws;
    size_t off = 0;
    auto alloc = [&](size_t bytes) -> char* {
        char* p = ws + off;
        off += (bytes + 255) & ~(size_t)255;
        return p;
    };
    float* h      = (float*)alloc((size_t)N_NODES * HID * 4);
    bf16*  hb     = (bf16*) alloc((size_t)N_NODES * HID * 2);
    float* hw     = (float*)alloc((size_t)N_NODES * HC * 4);
    float* accum  = (float*)alloc((size_t)N_NODES * HC * 4);
    float* al_s   = (float*)alloc((size_t)N_NODES * HEADS * 4);
    float* al_d   = (float*)alloc((size_t)N_NODES * HEADS * 4);
    float* emax   = (float*)alloc((size_t)N_NODES * HEADS * 4);
    float* denom  = (float*)alloc((size_t)N_NODES * HEADS * 4);
    float* ex     = (float*)alloc((size_t)ET * HEADS * 4);
    bf16*  Wt     = (bf16*) alloc((size_t)HC * HID * 2);
    float* pooled = (float*)alloc((size_t)G_GR * HID * 4);
    float* cnt    = (float*)alloc((size_t)G_GR * 4);

    const int TPB = 256;
    auto blocks = [](long long n, int t) { return (int)((n + t - 1) / t); };

    // input projection
    proj_kernel<<<blocks((long long)N_NODES * HID, TPB), TPB, 0, stream>>>(x, Wp, bp, h, hb);

    for (int l = 0; l < 3; ++l) {
        const float* W  = (const float*)d_in[7 + 4 * l];
        const float* as_ = (const float*)d_in[8 + 4 * l];
        const float* ad_ = (const float*)d_in[9 + 4 * l];
        const float* b   = (const float*)d_in[10 + 4 * l];

        wt_kernel<<<blocks(HID * HC, TPB), TPB, 0, stream>>>(W, Wt);

        dim3 gg(N_NODES / 16, 4);
        gemm_kernel<<<gg, 128, 0, stream>>>(hb, Wt, hw);

        logits_kernel<<<blocks((long long)N_NODES * HEADS, TPB), TPB, 0, stream>>>(hw, as_, ad_, al_s, al_d);

        fill_u32_kernel<<<blocks(N_NODES * HEADS, TPB), TPB, 0, stream>>>((uint32_t*)emax, 0xFF800000u, N_NODES * HEADS);
        fill_u32_kernel<<<blocks(N_NODES * HEADS, TPB), TPB, 0, stream>>>((uint32_t*)denom, 0u, N_NODES * HEADS);
        fill_u32_kernel<<<blocks((long long)N_NODES * HC, TPB), TPB, 0, stream>>>((uint32_t*)accum, 0u, N_NODES * HC);

        edge_max_kernel<<<blocks(ET, TPB), TPB, 0, stream>>>(ei, al_s, al_d, emax);
        edge_exp_kernel<<<blocks(ET, TPB), TPB, 0, stream>>>(ei, al_s, al_d, emax, ex, denom);
        edge_scatter_kernel<<<blocks((long long)ET * 64, TPB), TPB, 0, stream>>>(ei, ex, denom, hw, accum);

        finalize_kernel<<<blocks((long long)N_NODES * HID, TPB), TPB, 0, stream>>>(accum, b, h, hb);
    }

    fill_u32_kernel<<<blocks(G_GR * HID, TPB), TPB, 0, stream>>>((uint32_t*)pooled, 0u, G_GR * HID);
    fill_u32_kernel<<<blocks(G_GR, TPB), TPB, 0, stream>>>((uint32_t*)cnt, 0u, G_GR);

    pool_kernel<<<blocks((long long)N_NODES * HID, TPB), TPB, 0, stream>>>(h, bat, pooled, cnt);
    out_kernel<<<blocks(G_GR * D_OUT, TPB), TPB, 0, stream>>>(pooled, cnt, Wo, bo, (float*)d_out);
}
